// FIRConvolution_71803263255210
// MI455X (gfx1250) — matrix operand order repaired and checked by reading, not compiled, not run on hardware
//
#include <hip/hip_runtime.h>

typedef __attribute__((ext_vector_type(16))) __bf16 v16bf;
typedef __attribute__((ext_vector_type(8)))  float  v8f;

namespace {
constexpr int kLx = 131072;      // samples per (b,c) signal
constexpr int kLh = 8192;        // FIR taps
constexpr int kLhPad = 8704;     // kLh + 512 zero pad (tail rows m>0 overrun)
constexpr int kXWin = 11872;     // reversed-x window halves per block
constexpr int kOutPerBlock = 4096;
constexpr int kThreads = 256;    // 8 wave32s
constexpr int kGroups = 64;      // B*C
constexpr int kRBias = 3679;     // reversed index: r = kRBias - (g - bt0)
}

// Split f32 into bf16 hi (exact truncation) + bf16 lo (residual).
__device__ __forceinline__ void split_bf16(float v, __bf16* hi, __bf16* lo) {
  unsigned int u = __float_as_uint(v) & 0xFFFF0000u;
  float fh = __uint_as_float(u);
  *hi = (__bf16)fh;          // exact: top 8 mantissa bits
  *lo = (__bf16)(v - fh);    // next ~8 mantissa bits
}

__global__ void fir_wmma_bf16x3(const float* __restrict__ x,
                                const float* __restrict__ h,
                                float* __restrict__ y) {
  extern __shared__ char smem_raw[];
  __bf16* sh_hi = (__bf16*)smem_raw;          // filter hi          [kLhPad]
  __bf16* sh_lo = sh_hi + kLhPad;             // filter lo          [kLhPad]
  __bf16* sx_hi = sh_lo + kLhPad;             // reversed input hi  [kXWin]
  __bf16* sx_lo = sx_hi + kXWin;              // reversed input lo  [kXWin]

  const int group = blockIdx.y;               // b*C + c
  const int bt0 = blockIdx.x * kOutPerBlock;  // first output of this block
  const float* xg = x + (size_t)group * kLx;
  const float* hg = h + (size_t)group * kLh;
  float* yg = y + (size_t)group * kLx + bt0;

  const int tid = threadIdx.x;

  // Stage filter, split into bf16 hi/lo; zero-pad tail.
  for (int i = tid; i < kLhPad; i += kThreads) {
    float v = (i < kLh) ? hg[i] : 0.0f;
    split_bf16(v, &sh_hi[i], &sh_lo[i]);
  }
  // Stage input window REVERSED: sx[i] holds x[bt0 + kRBias - i]; zero outside
  // [0, kLx) gives causal left-edge handling for free.
  for (int i = tid; i < kXWin; i += kThreads) {
    int g = bt0 + kRBias - i;
    float v = (g >= 0 && g < kLx) ? xg[g] : 0.0f;
    split_bf16(v, &sx_hi[i], &sx_lo[i]);
  }
  __syncthreads();

  const int lane = tid & 31;
  const int wave = tid >> 5;
  const int lm = lane & 15;      // A: row M,  B: col N,  C: col N
  const int laneHi = lane >> 4;  // lane group selector per WMMA VGPR layouts
  const int t0 = wave << 9;      // this wave's 512-output span (block-relative)

  v8f acc_a = {};                // outputs t0 + 32m + n        (n < 16)
  v8f acc_b = {};                // outputs t0 + 16 + 32m + n   (complement)

  // Per-lane constant address parts.
  const int hbase = (lm << 5) + (laneHi << 4);            // 32*M + 16*laneHi
  const int rbase = kRBias - t0 - lm + (laneHi << 4);     // + k makes B index

  for (int jq = 0; jq < 256; ++jq) {
    const int jb = jq << 5;  // tap block base, step 32

    // A[m,k] = h[jb + 32m + k]; contiguous k = 16*laneHi + e, one ascending 16-run.
    const int hb = jb + hbase;
    v16bf Ah, Al;
#pragma unroll
    for (int e = 0; e < 8; ++e) {
      Ah[e] = sh_hi[hb + e];
      Al[e] = sh_lo[hb + e];
      Ah[e + 8] = sh_hi[hb + 8 + e];
      Al[e + 8] = sh_lo[hb + 8 + e];
    }

    // B[k,n] = x[t0 + n - jb - k]; reversed staging => one ascending 16-run.
    // Chain b (outputs +16) is the same run shifted by -16.
    const int r0 = rbase + jb;
    v16bf Bh, Bl, Dh, Dl;
#pragma unroll
    for (int e = 0; e < 16; ++e) {
      Bh[e] = sx_hi[r0 + e];
      Bl[e] = sx_lo[r0 + e];
      Dh[e] = sx_hi[r0 - 16 + e];
      Dl[e] = sx_lo[r0 - 16 + e];
    }

    // bf16x3: hi*hi + hi*lo + lo*hi, all accumulated in f32.
    acc_a = __builtin_amdgcn_wmma_f32_16x16x32_bf16(false, Ah, false, Bh, (short)0, acc_a, false, false);
    acc_a = __builtin_amdgcn_wmma_f32_16x16x32_bf16(false, Ah, false, Bl, (short)0, acc_a, false, false);
    acc_a = __builtin_amdgcn_wmma_f32_16x16x32_bf16(false, Al, false, Bh, (short)0, acc_a, false, false);
    acc_b = __builtin_amdgcn_wmma_f32_16x16x32_bf16(false, Ah, false, Dh, (short)0, acc_b, false, false);
    acc_b = __builtin_amdgcn_wmma_f32_16x16x32_bf16(false, Ah, false, Dl, (short)0, acc_b, false, false);
    acc_b = __builtin_amdgcn_wmma_f32_16x16x32_bf16(false, Al, false, Dh, (short)0, acc_b, false, false);
  }

  // C/D layout: VGPR r -> M = r + 8*laneHi, N = lane%16.
#pragma unroll
  for (int r = 0; r < 8; ++r) {
    const int m = r + (laneHi << 3);
    yg[t0 + (m << 5) + lm] = acc_a[r];
    yg[t0 + 16 + (m << 5) + lm] = acc_b[r];
  }
}

extern "C" void kernel_launch(void* const* d_in, const int* in_sizes, int n_in,
                              void* d_out, int out_size, void* d_ws, size_t ws_size,
                              hipStream_t stream) {
  (void)in_sizes; (void)n_in; (void)out_size; (void)d_ws; (void)ws_size;
  const float* x = (const float*)d_in[0];  // [32,2,131072] f32
  const float* h = (const float*)d_in[1];  // [32,2,8192]   f32
  float* y = (float*)d_out;                // [32,2,131072] f32

  dim3 grid(kLx / kOutPerBlock, kGroups);  // 32 x 64 blocks
  size_t shmem = (size_t)(2 * kLhPad + 2 * kXWin) * sizeof(__bf16);  // ~82 KB (CDNA5 LDS)
  fir_wmma_bf16x3<<<grid, kThreads, shmem, stream>>>(x, h, y);
}
